// TrajectoryGenerator_39591008535101
// MI455X (gfx1250) — matrix operand order repaired and checked by reading, not compiled, not run on hardware
//
#include <hip/hip_runtime.h>
#include <hip/hip_bf16.h>

// ---------------------------------------------------------------------------
// Social-GAN TrajectoryGenerator forward on MI455X (gfx1250, wave32, WMMA).
// S=32 scenes, N=32 peds, B=1024, E=64, H=128, PRE=512, MLP_D=1024, T=12.
//
// - All GEMMs: v_wmma_f32_16x16x32_bf16 (bf16 in, f32 accumulate).
// - Pool layer 1 collapsed algebraically to a rank-2 update (M2 = Wp1a@Wpe).
// - Pool layer 2: per-scene fused kernel; scene hcontrib (64KB fp32) and
//   bf16 Wp2 (128KB) staged into WGP LDS with global_load_async_to_lds_b128
//   (ASYNCcnt DMA), reused 32x; max-pool via per-lane max + shfl_xor(16).
// - Workspace ~10.4 MB (fits trivially in 192MB L2).
// ---------------------------------------------------------------------------

typedef __attribute__((ext_vector_type(16))) __bf16 v16bf;
typedef __attribute__((ext_vector_type(8)))  float  v8f;

#define WMMA_BF16(a, b, c) \
  __builtin_amdgcn_wmma_f32_16x16x32_bf16(false, (a), false, (b), (short)0, (c), false, false)

// Async DMA of one 16-byte chunk per lane: global -> LDS (ASYNCcnt-tracked).
__device__ inline void async_copy_b128(void* lds_dst, const void* gsrc) {
  const unsigned lds = (unsigned)(uintptr_t)lds_dst;                 // low 32b = LDS offset
  const unsigned long long ga = (unsigned long long)(uintptr_t)gsrc;
  asm volatile("global_load_async_to_lds_b128 %0, %1, off"
               :: "v"(lds), "v"(ga) : "memory");
}
__device__ inline void wait_async_all() {
  asm volatile("s_wait_asynccnt 0" ::: "memory");
}

// A fragment: 16x32 bf16 tile of row-major fp32 matrix A (lda), origin (m0,k0).
__device__ inline v16bf load_a_frag(const float* __restrict__ A, int lda, int m0, int k0) {
  const int lane = threadIdx.x & 31;
  const int m  = m0 + (lane & 15);
  const int hb = (lane >> 4) << 3;          // 0 or 8
  v16bf a;
#pragma unroll
  for (int e = 0; e < 16; ++e) {
    const int k = k0 + (hb << 1) + e;
    a[e] = (__bf16)A[m * lda + k];
  }
  return a;
}

// B fragment: 32x16 bf16 tile of W^T where W is row-major (N,K) fp32 (ldw).
__device__ inline v16bf load_b_fragT(const float* __restrict__ W, int ldw, int n0, int k0) {
  const int lane = threadIdx.x & 31;
  const int n  = n0 + (lane & 15);
  const int kb = k0 + ((lane >> 4) << 4);   // 0 or 16
  const float* wp = W + n * ldw + kb;
  v16bf b;
#pragma unroll
  for (int e = 0; e < 16; ++e) b[e] = (__bf16)wp[e];
  return b;
}

// ---------------------------------------------------------------------------
// Generic WMMA linear: Out = act( A1@W1^T + A2@W2^T + b1 + b2 )
// Grid: (M/32, N/64), block 128 (4 waves). Each wave: 32x16 output
// (2 accumulators) so each B fragment feeds two WMMAs.
// ---------------------------------------------------------------------------
__global__ __launch_bounds__(128) void k_gemm2(
    const float* __restrict__ A1, int lda1, const float* __restrict__ W1, int ldw1, int K1,
    const float* __restrict__ A2, int lda2, const float* __restrict__ W2, int ldw2, int K2,
    const float* __restrict__ b1, const float* __restrict__ b2,
    float* __restrict__ Out, int ldo, int relu) {
  const int lane = threadIdx.x & 31;
  const int wave = threadIdx.x >> 5;
  const int m0 = blockIdx.x * 32;
  const int n0 = blockIdx.y * 64 + wave * 16;

  v8f acc0 = {}, acc1 = {};
  for (int k = 0; k < K1; k += 32) {
    if (k + 32 < K1) {
      __builtin_prefetch(&A1[(m0 + (lane & 15)) * lda1 + k + 32], 0, 0);
    }
    v16bf b  = load_b_fragT(W1, ldw1, n0, k);
    v16bf a0 = load_a_frag(A1, lda1, m0, k);
    v16bf a1 = load_a_frag(A1, lda1, m0 + 16, k);
    acc0 = WMMA_BF16(a0, b, acc0);
    acc1 = WMMA_BF16(a1, b, acc1);
  }
  for (int k = 0; k < K2; k += 32) {
    v16bf b  = load_b_fragT(W2, ldw2, n0, k);
    v16bf a0 = load_a_frag(A2, lda2, m0, k);
    v16bf a1 = load_a_frag(A2, lda2, m0 + 16, k);
    acc0 = WMMA_BF16(a0, b, acc0);
    acc1 = WMMA_BF16(a1, b, acc1);
  }

  const int n  = n0 + (lane & 15);
  const int mb = m0 + ((lane >> 4) << 3);
  float bias = (b1 ? b1[n] : 0.f) + (b2 ? b2[n] : 0.f);
#pragma unroll
  for (int r = 0; r < 8; ++r) {
    float v0 = acc0[r] + bias;
    float v1 = acc1[r] + bias;
    if (relu) { v0 = fmaxf(v0, 0.f); v1 = fmaxf(v1, 0.f); }
    Out[(mb + r) * ldo + n]      = v0;
    Out[(mb + 16 + r) * ldo + n] = v1;
  }
}

// ---------------------------------------------------------------------------
// Once-per-launch: M2 = Wp1[:, :64] @ Wpe (512x2), bvec = bp1 + Wp1[:, :64]@bpe
// ---------------------------------------------------------------------------
__global__ __launch_bounds__(256) void k_prep(
    const float* __restrict__ Wp1, const float* __restrict__ Wpe,
    const float* __restrict__ bpe, const float* __restrict__ bp1,
    float* __restrict__ M2, float* __restrict__ bvec) {
  const int p = blockIdx.x * blockDim.x + threadIdx.x;
  if (p >= 512) return;
  float m0 = 0.f, m1 = 0.f, bv = bp1[p];
  for (int e = 0; e < 64; ++e) {
    const float w = Wp1[p * 192 + e];
    m0 += w * Wpe[e * 2];
    m1 += w * Wpe[e * 2 + 1];
    bv += w * bpe[e];
  }
  M2[p * 2] = m0; M2[p * 2 + 1] = m1; bvec[p] = bv;
}

// Once-per-launch: Wp2 (128,512) fp32 -> bf16 (async-LDS friendly layout).
__global__ __launch_bounds__(256) void k_prep_w(
    const float* __restrict__ Wp2, __bf16* __restrict__ out) {
  const int i = blockIdx.x * 256 + threadIdx.x;   // 65536 elements
  out[i] = (__bf16)Wp2[i];
}

// x0 = last_pos_rel @ Wsp^T + bsp ; lp = last_pos
__global__ __launch_bounds__(64) void k_embed(
    const float* __restrict__ lpr, const float* __restrict__ lastpos,
    const float* __restrict__ Wsp, const float* __restrict__ bsp,
    float* __restrict__ x, float* __restrict__ lp) {
  const int b = blockIdx.x, e = threadIdx.x;
  const float r0 = lpr[b * 2], r1 = lpr[b * 2 + 1];
  x[b * 64 + e] = r0 * Wsp[e * 2] + r1 * Wsp[e * 2 + 1] + bsp[e];
  if (e < 2) lp[b * 2 + e] = lastpos[b * 2 + e];
}

// LSTM elementwise: gates (B,512) in torch order i,f,g,o -> new h,c
__global__ __launch_bounds__(256) void k_lstm_elem(
    const float* __restrict__ gates, float* __restrict__ h, float* __restrict__ c) {
  const int idx = blockIdx.x * blockDim.x + threadIdx.x;  // B*H
  const int b = idx >> 7, j = idx & 127;
  const float* g = gates + b * 512;
  const float si = 1.f / (1.f + __expf(-g[j]));
  const float sf = 1.f / (1.f + __expf(-g[128 + j]));
  const float gg = tanhf(g[256 + j]);
  const float so = 1.f / (1.f + __expf(-g[384 + j]));
  const float c2 = sf * c[idx] + si * gg;
  c[idx] = c2;
  h[idx] = so * tanhf(c2);
}

// rel = h@Wh2p^T + bh2p ; out rel ; lp += rel (cur) ; x = rel@Wsp^T + bsp
__global__ __launch_bounds__(256) void k_h2p(
    const float* __restrict__ h, const float* __restrict__ Wh2p, const float* __restrict__ bh2p,
    const float* __restrict__ Wsp, const float* __restrict__ bsp,
    float* __restrict__ lp, float* __restrict__ x, float* __restrict__ outrel) {
  const int b = blockIdx.x * blockDim.x + threadIdx.x;
  if (b >= 1024) return;
  const float* hr = h + b * 128;
  float r0 = bh2p[0], r1 = bh2p[1];
  for (int k = 0; k < 128; ++k) {
    const float hv = hr[k];
    r0 += hv * Wh2p[k];
    r1 += hv * Wh2p[128 + k];
  }
  outrel[b * 2] = r0; outrel[b * 2 + 1] = r1;
  lp[b * 2] += r0;    lp[b * 2 + 1] += r1;   // cur = rel + lp
  for (int e = 0; e < 64; ++e)
    x[b * 64 + e] = r0 * Wsp[e * 2] + r1 * Wsp[e * 2 + 1] + bsp[e];
}

// ---------------------------------------------------------------------------
// Fused pool: per scene, build f1 = relu(hcontrib[j] + rel(i,j)·M2^T + bvec)
// on-the-fly in A fragments, GEMM against bf16 Wp2 (async-staged in LDS),
// max-pool over j, relu. One block per scene, 8 waves.
// LDS: hc 64KB fp32 + M2/bvec/pos ~6.3KB + Wp2 bf16 128KB ~= 199KB < 320KB/WGP.
// ---------------------------------------------------------------------------
__global__ __launch_bounds__(256) void k_pool2(
    const float*  __restrict__ hc,    // (1024,512) hcontrib
    const float*  __restrict__ M2,    // (512,2)
    const float*  __restrict__ bvec,  // (512)
    const float*  __restrict__ cur,   // (1024,2)
    const __bf16* __restrict__ Wp2b,  // (128,512) bf16
    const float*  __restrict__ bp2,   // (128)
    float* __restrict__ ph) {         // (1024,128)
  extern __shared__ char smem[];
  float*  hc_s  = (float*)smem;              // 32*512 fp32
  float*  m2_s  = hc_s + 32 * 512;           // 1024
  float*  bv_s  = m2_s + 1024;               // 512
  float*  pos_s = bv_s + 512;                // 64
  __bf16* w_s   = (__bf16*)(pos_s + 64);     // 128*512 bf16

  const int s = blockIdx.x, tid = threadIdx.x;

  // Async DMA staging (ASYNCcnt): scene hcontrib (4096 x b128), Wp2 (8192 x b128)
  const float* hc_g = hc + (size_t)s * 32 * 512;
  for (int c = tid; c < 4096; c += 256)
    async_copy_b128(hc_s + c * 4, hc_g + c * 4);
  for (int c = tid; c < 8192; c += 256)
    async_copy_b128(w_s + c * 8, Wp2b + c * 8);
  // Small tables via normal path
  for (int idx = tid; idx < 1024; idx += 256) m2_s[idx] = M2[idx];
  for (int idx = tid; idx < 512; idx += 256)  bv_s[idx] = bvec[idx];
  if (tid < 64)                               pos_s[tid] = cur[s * 64 + tid];
  wait_async_all();
  __syncthreads();

  const int lane = tid & 31, wave = tid >> 5;
  const int n0 = wave * 16;            // wave owns a 16-col slab of 128 outputs
  const int nn = lane & 15;
  const int hi = lane >> 4;

  for (int i = 0; i < 32; ++i) {
    const float pix = pos_s[i * 2], piy = pos_s[i * 2 + 1];
    v8f acc0 = {}, acc1 = {};
    for (int kk = 0; kk < 512; kk += 32) {
      // B fragment from LDS-resident bf16 Wp2
      v16bf bfrag;
      {
        const __bf16* wp = w_s + (n0 + nn) * 512 + kk + (hi << 4);
#pragma unroll
        for (int e = 0; e < 16; ++e) bfrag[e] = wp[e];
      }
      // A fragments (j-tiles 0..15 and 16..31): rank-2 update + relu, fused
#pragma unroll
      for (int mt = 0; mt < 2; ++mt) {
        const int j  = mt * 16 + nn;
        const float rx = pos_s[j * 2]     - pix;   // pos_j - pos_i
        const float ry = pos_s[j * 2 + 1] - piy;
        const float* hr = hc_s + j * 512;
        v16bf afrag;
#pragma unroll
        for (int e = 0; e < 16; ++e) {
          const int k = kk + (hi << 4) + e;
          const float v = hr[k] + rx * m2_s[k * 2] + ry * m2_s[k * 2 + 1] + bv_s[k];
          afrag[e] = (__bf16)fmaxf(v, 0.f);
        }
        if (mt == 0) acc0 = WMMA_BF16(afrag, bfrag, acc0);
        else         acc1 = WMMA_BF16(afrag, bfrag, acc1);
      }
    }
    // max over the 16 j-rows held by this lane, then combine lane halves
    float mx = acc0[0];
#pragma unroll
    for (int r = 0; r < 8; ++r) { mx = fmaxf(mx, acc0[r]); mx = fmaxf(mx, acc1[r]); }
    mx = fmaxf(mx, __shfl_xor(mx, 16, 32));
    if (hi == 0)
      ph[(s * 32 + i) * 128 + n0 + nn] = fmaxf(mx + bp2[n0 + nn], 0.f);
  }
}

// ---------------------------------------------------------------------------
extern "C" void kernel_launch(void* const* d_in, const int* in_sizes, int n_in,
                              void* d_out, int out_size, void* d_ws, size_t ws_size,
                              hipStream_t stream) {
  constexpr int B = 1024, H = 128, E = 64, PRE = 512, MLPD = 1024, T = 12;

  const float* last_pos     = (const float*)d_in[0];
  const float* last_pos_rel = (const float*)d_in[1];
  const float* h0   = (const float*)d_in[2];
  const float* c0   = (const float*)d_in[3];
  // d_in[4] seq_start_end (equal-size scenes, unused), d_in[5] seq_len (=12)
  const float* Wsp  = (const float*)d_in[6];
  const float* bsp  = (const float*)d_in[7];
  const float* Wih  = (const float*)d_in[8];
  const float* Whh  = (const float*)d_in[9];
  const float* bih  = (const float*)d_in[10];
  const float* bhh  = (const float*)d_in[11];
  const float* Wh2p = (const float*)d_in[12];
  const float* bh2p = (const float*)d_in[13];
  const float* Wpe  = (const float*)d_in[14];
  const float* bpe  = (const float*)d_in[15];
  const float* Wp1  = (const float*)d_in[16];
  const float* bp1  = (const float*)d_in[17];
  const float* Wp2  = (const float*)d_in[18];
  const float* bp2  = (const float*)d_in[19];
  const float* Wm1  = (const float*)d_in[20];
  const float* bm1  = (const float*)d_in[21];
  const float* Wm2  = (const float*)d_in[22];
  const float* bm2  = (const float*)d_in[23];

  // workspace layout (fp32 units)
  float* p = (float*)d_ws;
  float* hbuf  = p; p += B * H;      // LSTM / refreshed hidden
  float* cbuf  = p; p += B * H;      // cell state
  float* xbuf  = p; p += B * E;      // spatial embedding
  float* lpbuf = p; p += B * 2;      // running positions (cur)
  float* gates = p; p += B * 512;    // LSTM gates
  float* hcon  = p; p += B * PRE;    // pool hcontrib
  float* phbuf = p; p += B * H;      // pooled hidden
  float* dh1   = p; p += B * MLPD;   // MLP hidden
  float* M2b   = p; p += 1024;       // fused rank-2 matrix (512x2)
  float* bvb   = p; p += 512;        // fused bias
  __bf16* wbf  = (__bf16*)p;         // Wp2 in bf16 (65536 elems = 32768 floats)

  hipMemcpyAsync(hbuf, h0, (size_t)B * H * sizeof(float), hipMemcpyDeviceToDevice, stream);
  hipMemcpyAsync(cbuf, c0, (size_t)B * H * sizeof(float), hipMemcpyDeviceToDevice, stream);

  k_prep  <<<2, 256, 0, stream>>>(Wp1, Wpe, bpe, bp1, M2b, bvb);
  k_prep_w<<<256, 256, 0, stream>>>(Wp2, wbf);
  k_embed <<<B, 64, 0, stream>>>(last_pos_rel, last_pos, Wsp, bsp, xbuf, lpbuf);

  const size_t pool_smem = (size_t)(32 * 512 + 1024 + 512 + 64) * 4 + (size_t)128 * 512 * 2;
  float* out = (float*)d_out;

  for (int t = 0; t < T; ++t) {
    // gates = x@Wih^T + h@Whh^T + bih + bhh     (M=1024, N=512, K=64+128)
    k_gemm2<<<dim3(B / 32, 512 / 64), 128, 0, stream>>>(
        xbuf, E, Wih, E, E, hbuf, H, Whh, H, H, bih, bhh, gates, 512, 0);
    k_lstm_elem<<<(B * H) / 256, 256, 0, stream>>>(gates, hbuf, cbuf);

    // rel, cur, next x; write rels[t]
    k_h2p<<<B / 256, 256, 0, stream>>>(hbuf, Wh2p, bh2p, Wsp, bsp,
                                       lpbuf, xbuf, out + (size_t)t * B * 2);

    // hcontrib = h @ Wp1[:,64:]^T + bvec       (M=1024, N=512, K=128)
    k_gemm2<<<dim3(B / 32, PRE / 64), 128, 0, stream>>>(
        hbuf, H, Wp1 + 64, 192, H, nullptr, 0, nullptr, 0, 0, bvb, nullptr, hcon, PRE, 0);

    // fused rank-2 + relu + GEMM2 + max-pool per scene
    k_pool2<<<32, 256, pool_smem, stream>>>(hcon, M2b, bvb, lpbuf, wbf, bp2, phbuf);

    // dh1 = relu([h|ph] @ Wm1^T + bm1)          (M=1024, N=1024, K=128+128)
    k_gemm2<<<dim3(B / 32, MLPD / 64), 128, 0, stream>>>(
        hbuf, H, Wm1, 256, H, phbuf, H, Wm1 + 128, 256, H, bm1, nullptr, dh1, MLPD, 1);

    // h = relu(dh1 @ Wm2^T + bm2)               (M=1024, N=128, K=1024)
    k_gemm2<<<dim3(B / 32, H / 64), 128, 0, stream>>>(
        dh1, MLPD, Wm2, MLPD, MLPD, nullptr, 0, nullptr, 0, 0, bm2, nullptr, hbuf, H, 1);
  }
}
